// MultiHeadSelfAttention_4286377361601
// MI455X (gfx1250) — matrix operand order repaired and checked by reading, not compiled, not run on hardware
//
#include <hip/hip_runtime.h>
#include <hip/hip_bf16.h>
#include <math.h>

typedef __attribute__((ext_vector_type(16))) __bf16 v16bf;
typedef __attribute__((ext_vector_type(8)))  float  v8f;

#define T_SEQ   2048
#define C_DIM   1024
#define H_NUM   16
#define D_HEAD  64
#define B_BATCH 2
#define BT      (B_BATCH * T_SEQ)   // 4096
#define N_QKV   (3 * C_DIM)         // 3072

#if __has_builtin(__builtin_amdgcn_global_load_async_to_lds_b128)
#define GEMM_ASYNC 1
#else
#define GEMM_ASYNC 0
#endif

typedef int v4i __attribute__((vector_size(16)));
typedef __attribute__((address_space(1))) v4i GV4;   // global int4
typedef __attribute__((address_space(3))) v4i LV4;   // LDS int4

__device__ __forceinline__ void wait_async0() {
#if GEMM_ASYNC
#if __has_builtin(__builtin_amdgcn_s_wait_asynccnt)
  __builtin_amdgcn_s_wait_asynccnt(0);
#else
  asm volatile("s_wait_asynccnt 0" ::: "memory");
#endif
#endif
}

__device__ __forceinline__ void async_copy_b128(const void* g, void* l) {
#if GEMM_ASYNC
  __builtin_amdgcn_global_load_async_to_lds_b128(
      (GV4*)(unsigned long long)g,
      (LV4*)(unsigned int)(unsigned long long)l, 0, 0);
#else
  (void)g; (void)l;
#endif
}

__device__ __forceinline__ unsigned short f2bf(float x) {
  unsigned int u = __float_as_uint(x);
  u += 0x7FFFu + ((u >> 16) & 1u);        // round-to-nearest-even
  return (unsigned short)(u >> 16);
}

union FragB16 { v16bf v; unsigned int u[8]; };

// A fragment: 16x32 bf16, row-major source with row stride ldr (elements).
// Contiguous k order (matches load_frag_b): lane row = lane%16; dword j: K = (lane/16)*16 + 2j.
__device__ __forceinline__ v16bf load_frag_a(const unsigned short* base, int ldr) {
  const int lane = threadIdx.x & 31;
  const int m = lane & 15, hh = lane >> 4;
  const unsigned short* row = base + (size_t)m * ldr;
  FragB16 f;
#pragma unroll
  for (int j = 0; j < 4; ++j) {
    f.u[j]     = *(const unsigned int*)(row + hh * 16 + 2 * j);
    f.u[4 + j] = *(const unsigned int*)(row + hh * 16 + 8 + 2 * j);
  }
  return f.v;
}

// B fragment: 32x16 bf16. Source "k-major": element (k,n) at base + n*ldr + k.
// Contiguous k order (matches load_frag_a): lane col = lane%16; dword j: K = (lane/16)*16 + 2j.
__device__ __forceinline__ v16bf load_frag_b(const unsigned short* base, int ldr) {
  const int lane = threadIdx.x & 31;
  const int n = lane & 15, hh = lane >> 4;
  const unsigned short* col = base + (size_t)n * ldr + hh * 16;
  FragB16 f;
#pragma unroll
  for (int j = 0; j < 8; ++j) f.u[j] = *(const unsigned int*)(col + 2 * j);
  return f.v;
}

__device__ __forceinline__ v8f wmma_bf16(v16bf a, v16bf b, v8f c) {
  return __builtin_amdgcn_wmma_f32_16x16x32_bf16(false, a, false, b, (short)0, c,
                                                 false, false);
}

__device__ __forceinline__ float redmax16(float v) {
#pragma unroll
  for (int m = 1; m < 16; m <<= 1) v = fmaxf(v, __shfl_xor(v, m, 32));
  return v;   // reduces within each 16-lane half (masks < 16 never cross halves)
}
__device__ __forceinline__ float redsum16(float v) {
#pragma unroll
  for (int m = 1; m < 16; m <<= 1) v += __shfl_xor(v, m, 32);
  return v;
}

// ---------------------------------------------------------------------------
// fp32 -> bf16 conversion (plain)
// ---------------------------------------------------------------------------
__global__ void cvt_bf16_kernel(const float* __restrict__ src,
                                unsigned short* __restrict__ dst, int n) {
  int i = blockIdx.x * blockDim.x + threadIdx.x;
  if (i < n) dst[i] = f2bf(src[i]);
}

// ---------------------------------------------------------------------------
// fp32 -> bf16 conversion with transpose: src K x N row-major -> dst N x K.
// ---------------------------------------------------------------------------
__global__ __launch_bounds__(256) void cvt_transpose_kernel(
    const float* __restrict__ src, unsigned short* __restrict__ dst, int K, int N) {
  __shared__ float tile[32][33];
  const int k0 = blockIdx.y * 32, n0 = blockIdx.x * 32;
  const int tx = threadIdx.x & 31, ty = threadIdx.x >> 5;   // ty: 0..7
#pragma unroll
  for (int i = ty; i < 32; i += 8)
    tile[i][tx] = src[(size_t)(k0 + i) * N + n0 + tx];
  __syncthreads();
#pragma unroll
  for (int i = ty; i < 32; i += 8)
    dst[(size_t)(n0 + i) * K + k0 + tx] = f2bf(tile[tx][i]);
}

// ---------------------------------------------------------------------------
// Shared GEMM mainloop. A: M x K row-major bf16. Bt: N x K row-major bf16.
// 128x128 macro tile, 8 waves (4M x 2N), wave tile 32x64, K chunk 32.
// Double-buffered LDS. Async path: GLOBAL_LOAD_ASYNC_TO_LDS_B128 copies the
// next K-tile straight into LDS (no staging VGPRs); each wave does
// s_wait_asynccnt 0 then barrier before consuming, and copies into `nxt` are
// issued only after the barrier that retires all reads of `nxt`.
// ---------------------------------------------------------------------------
__device__ __forceinline__ void gemm_tiles(const unsigned short* __restrict__ A,
                                           const unsigned short* __restrict__ Bt,
                                           int K, int rowBase, int colBase,
                                           unsigned short* As, unsigned short* Bs,
                                           v8f (&acc)[2][4]) {
  const int tid = threadIdx.x;
  const int wid = tid >> 5;
  const int wm = wid & 3, wn = wid >> 2;
  const int nk = K / 32;

  // staging chunk coords for this thread (2 chunks per tile per array)
  int srow[2], skc[2];
#pragma unroll
  for (int i = 0; i < 2; ++i) {
    int c = tid + i * 256;              // 0..511
    srow[i] = c >> 2;
    skc[i] = (c & 3) * 8;
  }

#if GEMM_ASYNC
  // prologue: async-copy tile 0 into buffer 0
#pragma unroll
  for (int i = 0; i < 2; ++i) {
    async_copy_b128(A  + (size_t)(rowBase + srow[i]) * K + skc[i],
                    As + srow[i] * 40 + skc[i]);
    async_copy_b128(Bt + (size_t)(colBase + srow[i]) * K + skc[i],
                    Bs + srow[i] * 40 + skc[i]);
  }
  for (int kt = 0; kt < nk; ++kt) {
    const int cur = kt & 1, nxt = cur ^ 1;
    wait_async0();        // this wave's copies into `cur` have landed
    __syncthreads();      // all waves' copies landed; all reads of `nxt` retired
    if (kt + 1 < nk) {    // overlap next tile's DMA with current compute
      const int k0 = (kt + 1) * 32;
      unsigned short* Asn = As + nxt * (128 * 40);
      unsigned short* Bsn = Bs + nxt * (128 * 40);
#pragma unroll
      for (int i = 0; i < 2; ++i) {
        async_copy_b128(A  + (size_t)(rowBase + srow[i]) * K + k0 + skc[i],
                        Asn + srow[i] * 40 + skc[i]);
        async_copy_b128(Bt + (size_t)(colBase + srow[i]) * K + k0 + skc[i],
                        Bsn + srow[i] * 40 + skc[i]);
      }
    }
    unsigned short* Asc = As + cur * (128 * 40);
    unsigned short* Bsc = Bs + cur * (128 * 40);
    v16bf afr[2], bfr[4];
#pragma unroll
    for (int fi = 0; fi < 2; ++fi)
      afr[fi] = load_frag_a(Asc + (wm * 32 + fi * 16) * 40, 40);
#pragma unroll
    for (int fj = 0; fj < 4; ++fj)
      bfr[fj] = load_frag_b(Bsc + (wn * 64 + fj * 16) * 40, 40);
#pragma unroll
    for (int fi = 0; fi < 2; ++fi)
#pragma unroll
      for (int fj = 0; fj < 4; ++fj)
        acc[fi][fj] = wmma_bf16(afr[fi], bfr[fj], acc[fi][fj]);
  }
#else
  uint4 ra[2], rb[2];
  // prologue: stage tile 0 into buffer 0
#pragma unroll
  for (int i = 0; i < 2; ++i) {
    ra[i] = *(const uint4*)(A  + (size_t)(rowBase + srow[i]) * K + skc[i]);
    rb[i] = *(const uint4*)(Bt + (size_t)(colBase + srow[i]) * K + skc[i]);
  }
#pragma unroll
  for (int i = 0; i < 2; ++i) {
    *(uint4*)(As + srow[i] * 40 + skc[i]) = ra[i];
    *(uint4*)(Bs + srow[i] * 40 + skc[i]) = rb[i];
  }
  for (int kt = 0; kt < nk; ++kt) {
    const int cur = kt & 1, nxt = cur ^ 1;
    unsigned short* Asc = As + cur * (128 * 40);
    unsigned short* Bsc = Bs + cur * (128 * 40);
    if (kt + 1 < nk) {
      const int k0 = (kt + 1) * 32;
#pragma unroll
      for (int i = 0; i < 2; ++i) {
        ra[i] = *(const uint4*)(A  + (size_t)(rowBase + srow[i]) * K + k0 + skc[i]);
        rb[i] = *(const uint4*)(Bt + (size_t)(colBase + srow[i]) * K + k0 + skc[i]);
      }
      if (kt + 2 < nk) {
        __builtin_prefetch(A + (size_t)(rowBase + (tid >> 1)) * K + (kt + 2) * 32, 0, 0);
        __builtin_prefetch(Bt + (size_t)(colBase + (tid >> 1)) * K + (kt + 2) * 32, 0, 0);
      }
    }
    __syncthreads();
    v16bf afr[2], bfr[4];
#pragma unroll
    for (int fi = 0; fi < 2; ++fi)
      afr[fi] = load_frag_a(Asc + (wm * 32 + fi * 16) * 40, 40);
#pragma unroll
    for (int fj = 0; fj < 4; ++fj)
      bfr[fj] = load_frag_b(Bsc + (wn * 64 + fj * 16) * 40, 40);
#pragma unroll
    for (int fi = 0; fi < 2; ++fi)
#pragma unroll
      for (int fj = 0; fj < 4; ++fj)
        acc[fi][fj] = wmma_bf16(afr[fi], bfr[fj], acc[fi][fj]);
    if (kt + 1 < nk) {
      unsigned short* Asn = As + nxt * (128 * 40);
      unsigned short* Bsn = Bs + nxt * (128 * 40);
#pragma unroll
      for (int i = 0; i < 2; ++i) {
        *(uint4*)(Asn + srow[i] * 40 + skc[i]) = ra[i];
        *(uint4*)(Bsn + srow[i] * 40 + skc[i]) = rb[i];
      }
    }
  }
#endif
}

// ---------------------------------------------------------------------------
// Stage 2: qkv = x @ w_qkv, fused RoPE epilogue, scatter Q[B,H,T,D],
// K[B,H,T,D], V transposed -> Vt[B,H,D,T]
// ---------------------------------------------------------------------------
__global__ __launch_bounds__(256)
__attribute__((amdgpu_waves_per_eu(2))) void qkv_gemm_kernel(
    const unsigned short* __restrict__ xb, const unsigned short* __restrict__ wqkvT,
    unsigned short* __restrict__ Qb, unsigned short* __restrict__ Kb,
    unsigned short* __restrict__ Vtb) {
  __shared__ __align__(16) unsigned short As[2 * 128 * 40];
  __shared__ __align__(16) unsigned short Bs[2 * 128 * 40];
  v8f acc[2][4];
  const v8f z = {0.f, 0.f, 0.f, 0.f, 0.f, 0.f, 0.f, 0.f};
#pragma unroll
  for (int fi = 0; fi < 2; ++fi)
#pragma unroll
    for (int fj = 0; fj < 4; ++fj) acc[fi][fj] = z;

  const int rowBase = blockIdx.y * 128, colBase = blockIdx.x * 128;
  gemm_tiles(xb, wqkvT, C_DIM, rowBase, colBase, As, Bs, acc);

  const int lane = threadIdx.x & 31;
  const int nlane = lane & 15, hh = lane >> 4;
  const int wid = threadIdx.x >> 5;
  const int wm = wid & 3, wn = wid >> 2;
#pragma unroll
  for (int fi = 0; fi < 2; ++fi)
#pragma unroll
    for (int fj = 0; fj < 4; ++fj)
#pragma unroll
      for (int r = 0; r < 8; ++r) {
        float v = acc[fi][fj][r];
        int mg = rowBase + wm * 32 + fi * 16 + r + 8 * hh;  // b*T + t
        int ng = colBase + wn * 64 + fj * 16 + nlane;       // 0..3071
        int sec = ng >> 10;                                 // 0=q 1=k 2=v
        int cidx = ng & 1023;
        int head = cidx >> 6, d = cidx & 63;
        int b = mg >> 11, t = mg & (T_SEQ - 1);
        if (sec < 2) {
          float vp = __shfl_xor(v, 1);    // partner column (even<->odd of pair)
          int i2 = d >> 1;
          float ang = (float)t * __expf(-0.2878231366242557f * (float)i2);
          float cs = __cosf(ang), sn = __sinf(ang);
          float o = (d & 1) ? (vp * sn + v * cs) : (v * cs - vp * sn);
          size_t idx = (((size_t)(b * H_NUM + head)) * T_SEQ + t) * D_HEAD + d;
          if (sec == 0) Qb[idx] = f2bf(o);
          else          Kb[idx] = f2bf(o);
        } else {
          Vtb[(((size_t)(b * H_NUM + head)) * D_HEAD + d) * T_SEQ + t] = f2bf(v);
        }
      }
}

// ---------------------------------------------------------------------------
// Stage 3: causal flash attention. grid=(T/64, B*H), block=128 (4 waves).
// ---------------------------------------------------------------------------
__global__ __launch_bounds__(128)
__attribute__((amdgpu_waves_per_eu(1))) void attn_kernel(
    const unsigned short* __restrict__ Qb, const unsigned short* __restrict__ Kb,
    const unsigned short* __restrict__ Vtb, unsigned short* __restrict__ Ob) {
  __shared__ __align__(16) unsigned short Pt[4 * 16 * 40];
  const int bh = blockIdx.y;                 // b*H + h
  const int b = bh >> 4, h = bh & 15;
  const int wid = threadIdx.x >> 5;
  const int lane = threadIdx.x & 31;
  const int nlane = lane & 15, hh = lane >> 4;
  const int qbase = blockIdx.x * 64 + wid * 16;
  unsigned short* pw = Pt + wid * 16 * 40;

  const unsigned short* Qh = Qb + ((size_t)bh * T_SEQ) * D_HEAD;
  const unsigned short* Kh = Kb + ((size_t)bh * T_SEQ) * D_HEAD;
  const unsigned short* Vh = Vtb + ((size_t)bh * D_HEAD) * T_SEQ;

  v16bf aq[2];
#pragma unroll
  for (int dc = 0; dc < 2; ++dc)
    aq[dc] = load_frag_a(Qh + (size_t)qbase * D_HEAD + dc * 32, D_HEAD);

  const v8f z = {0.f, 0.f, 0.f, 0.f, 0.f, 0.f, 0.f, 0.f};
  v8f oacc[4] = {z, z, z, z};
  float rmax[8], rsum[8];
#pragma unroll
  for (int r = 0; r < 8; ++r) { rmax[r] = -3.4028235e38f; rsum[r] = 0.f; }

  const int nch = (qbase + 47) / 32;   // covers keys 0..qbase+15
  for (int kc = 0; kc < nch; ++kc) {
    const int kt0 = kc * 32;
    v8f s0 = z, s1 = z;
#pragma unroll
    for (int dc = 0; dc < 2; ++dc) {
      v16bf b0 = load_frag_b(Kh + (size_t)kt0 * D_HEAD + dc * 32, D_HEAD);
      v16bf b1 = load_frag_b(Kh + (size_t)(kt0 + 16) * D_HEAD + dc * 32, D_HEAD);
      s0 = wmma_bf16(aq[dc], b0, s0);
      s1 = wmma_bf16(aq[dc], b1, s1);
    }
    // online softmax per row (row r+8*hh lives in VGPR r across a 16-lane half)
#pragma unroll
    for (int r = 0; r < 8; ++r) {
      int m = r + 8 * hh;
      int qi = qbase + m;
      float v0 = s0[r] * 0.125f;
      float v1 = s1[r] * 0.125f;
      if (kt0 + nlane > qi)      v0 = -3.4028235e38f;
      if (kt0 + 16 + nlane > qi) v1 = -3.4028235e38f;
      float cm = redmax16(fmaxf(v0, v1));
      float nm = fmaxf(rmax[r], cm);
      float alpha = __expf(rmax[r] - nm);
      float p0 = __expf(v0 - nm);
      float p1 = __expf(v1 - nm);
      float csum = redsum16(p0 + p1);
      rsum[r] = rsum[r] * alpha + csum;
      rmax[r] = nm;
#pragma unroll
      for (int jn = 0; jn < 4; ++jn) oacc[jn][r] *= alpha;
      pw[m * 40 + nlane]      = f2bf(p0);
      pw[m * 40 + 16 + nlane] = f2bf(p1);
    }
    asm volatile("" ::: "memory");   // LDS in-order within wave; block reordering
    v16bf pa = load_frag_a(pw, 40);
#pragma unroll
    for (int jn = 0; jn < 4; ++jn) {
      v16bf bv = load_frag_b(Vh + (size_t)(jn * 16) * T_SEQ + kt0, T_SEQ);
      oacc[jn] = wmma_bf16(pa, bv, oacc[jn]);
    }
  }
  // normalize and emit O as bf16 rows of the (BT x C) matrix for the out-proj
#pragma unroll
  for (int r = 0; r < 8; ++r) {
    int m = r + 8 * hh;
    float inv = 1.0f / rsum[r];
    size_t row = (size_t)(b * T_SEQ + qbase + m) * C_DIM + h * D_HEAD;
#pragma unroll
    for (int jn = 0; jn < 4; ++jn)
      Ob[row + jn * 16 + nlane] = f2bf(oacc[jn][r] * inv);
  }
}

// ---------------------------------------------------------------------------
// Stage 4: out = O @ w_out (fp32 result), w_out pre-transposed
// ---------------------------------------------------------------------------
__global__ __launch_bounds__(256)
__attribute__((amdgpu_waves_per_eu(2))) void out_gemm_kernel(
    const unsigned short* __restrict__ Ob, const unsigned short* __restrict__ woutT,
    float* __restrict__ out) {
  __shared__ __align__(16) unsigned short As[2 * 128 * 40];
  __shared__ __align__(16) unsigned short Bs[2 * 128 * 40];
  v8f acc[2][4];
  const v8f z = {0.f, 0.f, 0.f, 0.f, 0.f, 0.f, 0.f, 0.f};
#pragma unroll
  for (int fi = 0; fi < 2; ++fi)
#pragma unroll
    for (int fj = 0; fj < 4; ++fj) acc[fi][fj] = z;

  const int rowBase = blockIdx.y * 128, colBase = blockIdx.x * 128;
  gemm_tiles(Ob, woutT, C_DIM, rowBase, colBase, As, Bs, acc);

  const int lane = threadIdx.x & 31;
  const int nlane = lane & 15, hh = lane >> 4;
  const int wid = threadIdx.x >> 5;
  const int wm = wid & 3, wn = wid >> 2;
#pragma unroll
  for (int fi = 0; fi < 2; ++fi)
#pragma unroll
    for (int fj = 0; fj < 4; ++fj)
#pragma unroll
      for (int r = 0; r < 8; ++r) {
        int mg = rowBase + wm * 32 + fi * 16 + r + 8 * hh;
        int ng = colBase + wn * 64 + fj * 16 + nlane;
        out[(size_t)mg * C_DIM + ng] = acc[fi][fj][r];
      }
}

// ---------------------------------------------------------------------------
extern "C" void kernel_launch(void* const* d_in, const int* in_sizes, int n_in,
                              void* d_out, int out_size, void* d_ws, size_t ws_size,
                              hipStream_t stream) {
  (void)in_sizes; (void)n_in; (void)out_size; (void)ws_size;
  const float* x     = (const float*)d_in[0];
  const float* w_qkv = (const float*)d_in[1];
  const float* w_out = (const float*)d_in[2];
  float* out = (float*)d_out;

  char* ws = (char*)d_ws;
  size_t off = 0;
  const size_t M2 = (size_t)BT * C_DIM * 2;       // 8 MB each
  unsigned short* xb     = (unsigned short*)(ws + off); off += M2;
  unsigned short* wqkvT  = (unsigned short*)(ws + off); off += (size_t)C_DIM * N_QKV * 2;
  unsigned short* woutT  = (unsigned short*)(ws + off); off += (size_t)C_DIM * C_DIM * 2;
  unsigned short* Qb     = (unsigned short*)(ws + off); off += M2;
  unsigned short* Kb     = (unsigned short*)(ws + off); off += M2;
  unsigned short* Vtb    = (unsigned short*)(ws + off); off += M2;
  unsigned short* Ob     = (unsigned short*)(ws + off); off += M2;

  const int nx = BT * C_DIM;
  cvt_bf16_kernel<<<(nx + 255) / 256, 256, 0, stream>>>(x, xb, nx);
  // weights: convert + transpose (K x N -> N x K)
  cvt_transpose_kernel<<<dim3(N_QKV / 32, C_DIM / 32), 256, 0, stream>>>(
      w_qkv, wqkvT, C_DIM, N_QKV);
  cvt_transpose_kernel<<<dim3(C_DIM / 32, C_DIM / 32), 256, 0, stream>>>(
      w_out, woutT, C_DIM, C_DIM);

  qkv_gemm_kernel<<<dim3(N_QKV / 128, BT / 128), 256, 0, stream>>>(xb, wqkvT, Qb, Kb, Vtb);
  attn_kernel<<<dim3(T_SEQ / 64, B_BATCH * H_NUM), 128, 0, stream>>>(Qb, Kb, Vtb, Ob);
  out_gemm_kernel<<<dim3(C_DIM / 128, BT / 128), 256, 0, stream>>>(Ob, woutT, out);
}
